// CorticalColumn_16801912062743
// MI455X (gfx1250) — matrix operand order repaired and checked by reading, not compiled, not run on hardware
//
#include <hip/hip_runtime.h>
#include <hip/hip_bf16.h>

typedef __attribute__((ext_vector_type(16))) __bf16 v16bf;
typedef __attribute__((ext_vector_type(8)))  float  v8f;

#define NROWS 32768
#define DIM   2048
#define KSEL  1638
#define KPAD  1664           // 26 * 64, M padded to the 64-row wave tile
#define KT    (DIM / 32)     // 64 K-tiles of 32
#define NT16  (DIM / 16)     // 128 N-tiles of 16

// ---------------------------------------------------------------------------
// Pack W (row-major fp32 [DIM,DIM]) into bf16 in the wave32 WMMA B-operand
// layout: element e of lane L in tile (nt16, kt) = W[kt*32 + 8*(L>>4) + (e<8 ? e : e+8)][nt16*16 + (L&15)]
// Stored as 16 contiguous bf16 (32B) per (tile, lane).
// ---------------------------------------------------------------------------
__global__ void pack_w_kernel(const float* __restrict__ W, __bf16* __restrict__ Bpk) {
  int t = blockIdx.x * blockDim.x + threadIdx.x;   // NT16*KT*32 = 262144 threads
  int lane = t & 31;
  int kt   = (t >> 5) & (KT - 1);
  int nt   = t >> 11;
  if (nt >= NT16) return;
  int n     = nt * 16 + (lane & 15);
  int kbase = kt * 32 + ((lane >> 4) << 3);
  v16bf v;
#pragma unroll
  for (int e = 0; e < 16; ++e)
    v[e] = (__bf16)W[(size_t)(kbase + (e < 8 ? e : e + 8)) * DIM + n];
  *(v16bf*)(Bpk + (size_t)t * 16) = v;
}

// ---------------------------------------------------------------------------
__global__ void zero_out_kernel(float4* __restrict__ o, int n4) {
  int i = blockIdx.x * blockDim.x + threadIdx.x;
  int stride = gridDim.x * blockDim.x;
  float4 z = {0.f, 0.f, 0.f, 0.f};
  for (; i < n4; i += stride) o[i] = z;
}

// ---------------------------------------------------------------------------
// Gate scores: one wave per row, fp32 dot product, wave32 shfl reduction.
// ---------------------------------------------------------------------------
__global__ void gate_kernel(const float* __restrict__ x, const float* __restrict__ Wg,
                            const float* __restrict__ bgp, float* __restrict__ gate) {
  int wave = threadIdx.x >> 5;
  int lane = threadIdx.x & 31;
  int row  = blockIdx.x * 8 + wave;
  const float4* xr = (const float4*)(x + (size_t)row * DIM);
  const float4* wg = (const float4*)Wg;
  float acc = 0.f;
#pragma unroll
  for (int j = 0; j < DIM / 128; ++j) {
    float4 a = xr[j * 32 + lane];
    float4 b = wg[j * 32 + lane];
    acc += a.x * b.x + a.y * b.y + a.z * b.z + a.w * b.w;
  }
#pragma unroll
  for (int off = 16; off > 0; off >>= 1) acc += __shfl_xor(acc, off, 32);
  if (lane == 0) gate[row] = acc + bgp[0];
}

// ---------------------------------------------------------------------------
// Top-K: single workgroup bitwise radix-select on order-preserving uint keys.
// ---------------------------------------------------------------------------
__device__ inline unsigned fmap(float f) {
  unsigned u = __float_as_uint(f);
  return (u & 0x80000000u) ? ~u : (u | 0x80000000u);
}

__global__ void select_kernel(const float* __restrict__ gate, int* __restrict__ idx,
                              float* __restrict__ mask) {
  __shared__ unsigned warpSums[32];
  __shared__ unsigned totalSh;
  __shared__ unsigned cGT, cTie;
  const int tid = threadIdx.x;                 // 1024 threads
  const int lane = tid & 31, wid = tid >> 5;

  unsigned prefix = 0;
  for (int bit = 31; bit >= 0; --bit) {
    unsigned cand = prefix | (1u << bit);
    unsigned local = 0;
    for (int i = tid; i < NROWS; i += 1024)
      local += (fmap(gate[i]) >= cand) ? 1u : 0u;
#pragma unroll
    for (int off = 16; off > 0; off >>= 1) local += __shfl_xor(local, off, 32);
    if (lane == 0) warpSums[wid] = local;
    __syncthreads();
    if (tid == 0) {
      unsigned t = 0;
      for (int w = 0; w < 32; ++w) t += warpSums[w];
      totalSh = t;
    }
    __syncthreads();
    if (totalSh >= (unsigned)KSEL) prefix = cand;
    __syncthreads();
  }
  // count strictly-greater, reset compaction counters
  {
    unsigned local = 0;
    for (int i = tid; i < NROWS; i += 1024)
      local += (fmap(gate[i]) > prefix) ? 1u : 0u;
#pragma unroll
    for (int off = 16; off > 0; off >>= 1) local += __shfl_xor(local, off, 32);
    if (lane == 0) warpSums[wid] = local;
    __syncthreads();
    if (tid == 0) {
      unsigned t = 0;
      for (int w = 0; w < 32; ++w) t += warpSums[w];
      totalSh = t;
      cGT = 0; cTie = 0;
    }
    __syncthreads();
  }
  unsigned nG = totalSh;
  unsigned needTies = (unsigned)KSEL - nG;
  for (int i = tid; i < NROWS; i += 1024) {
    unsigned u = fmap(gate[i]);
    if (u > prefix) {
      unsigned p = atomicAdd(&cGT, 1u);
      idx[p] = i;
      mask[i] = 1.0f;
    } else if (u == prefix) {
      unsigned t = atomicAdd(&cTie, 1u);
      if (t < needTies) {
        idx[nG + t] = i;
        mask[i] = 1.0f;
      }
    }
  }
  __syncthreads();
  for (int i = KSEL + tid; i < KPAD; i += 1024) idx[i] = 0;   // valid padding rows
}

// ---------------------------------------------------------------------------
// Load one 16x32 A tile (fp32 row-major source, optional row indirection) as
// bf16 in the wave32 WMMA A layout:
//   lane L, element e -> row (L&15), k = k0 + (e<8 ? e : e+8) + 8*(L>>4)
// i.e. two contiguous 8-float runs per lane -> 4x b128 loads + cvt.
// ---------------------------------------------------------------------------
__device__ inline v16bf load_a_tile(const float* __restrict__ A, const int* __restrict__ idx,
                                    int m0, int k0, int lane) {
  int m = m0 + (lane & 15);
  int row = idx ? idx[m] : m;
  int kb = k0 + ((lane >> 4) << 3);
  const float* p = A + (size_t)row * DIM + kb;
  float4 f0 = *(const float4*)(p);
  float4 f1 = *(const float4*)(p + 4);
  float4 f2 = *(const float4*)(p + 16);
  float4 f3 = *(const float4*)(p + 20);
  v16bf a;
  a[0]  = (__bf16)f0.x; a[1]  = (__bf16)f0.y; a[2]  = (__bf16)f0.z; a[3]  = (__bf16)f0.w;
  a[4]  = (__bf16)f1.x; a[5]  = (__bf16)f1.y; a[6]  = (__bf16)f1.z; a[7]  = (__bf16)f1.w;
  a[8]  = (__bf16)f2.x; a[9]  = (__bf16)f2.y; a[10] = (__bf16)f2.z; a[11] = (__bf16)f2.w;
  a[12] = (__bf16)f3.x; a[13] = (__bf16)f3.y; a[14] = (__bf16)f3.z; a[15] = (__bf16)f3.w;
  return a;
}

// ---------------------------------------------------------------------------
// GEMM: each wave computes a 64x64 output tile with 4x4 v_wmma_f32_16x16x32_bf16
// per K-step.  MODE 0: A = x gathered via idx, epilogue bias+ReLU -> h.
//            MODE 1: A = h (identity), epilogue bias, scatter rows via idx.
// ---------------------------------------------------------------------------
template <int MODE>
__global__ void gemm_kernel(const float* __restrict__ A, const int* __restrict__ idx,
                            const __bf16* __restrict__ Bpk, const float* __restrict__ bias,
                            float* __restrict__ out) {
  const int lane = threadIdx.x & 31;
  const int wid  = blockIdx.x * (blockDim.x >> 5) + (threadIdx.x >> 5);
  const int mt = wid >> 5;           // 26 M-tiles of 64
  const int nt = wid & 31;           // 32 N-tiles of 64
  const int m0 = mt * 64;
  const int n0 = nt * 64;

  const v8f vzero = {0.f, 0.f, 0.f, 0.f, 0.f, 0.f, 0.f, 0.f};
  v8f acc[4][4];
#pragma unroll
  for (int i = 0; i < 4; ++i)
#pragma unroll
    for (int j = 0; j < 4; ++j) acc[i][j] = vzero;

  for (int kt = 0; kt < KT; ++kt) {
    const int k0 = kt * 32;
    v16bf a[4], b[4];
#pragma unroll
    for (int i = 0; i < 4; ++i)
      a[i] = load_a_tile(A, MODE == 0 ? idx : nullptr, m0 + i * 16, k0, lane);
#pragma unroll
    for (int j = 0; j < 4; ++j) {
      const int ntile = (n0 >> 4) + j;
      b[j] = *(const v16bf*)(Bpk + ((size_t)(ntile * KT + kt) * 32 + lane) * 16);
    }
#pragma unroll
    for (int i = 0; i < 4; ++i)
#pragma unroll
      for (int j = 0; j < 4; ++j)
        acc[i][j] = __builtin_amdgcn_wmma_f32_16x16x32_bf16(
            false, a[i], false, b[j], (short)0, acc[i][j], false, false);
  }

  // Epilogue: C layout -> lane L holds n = (L&15), rows r + 8*(L>>4)
#pragma unroll
  for (int i = 0; i < 4; ++i) {
#pragma unroll
    for (int j = 0; j < 4; ++j) {
#pragma unroll
      for (int r = 0; r < 8; ++r) {
        int gm = m0 + i * 16 + r + ((lane >> 4) << 3);
        int gn = n0 + j * 16 + (lane & 15);
        float v = acc[i][j][r] + bias[gn];
        if (MODE == 0) {
          v = v > 0.f ? v : 0.f;
          out[(size_t)gm * DIM + gn] = v;
        } else {
          if (gm < KSEL) {
            int row = idx[gm];
            out[(size_t)row * DIM + gn] = v;
          }
        }
      }
    }
  }
}

// ---------------------------------------------------------------------------
extern "C" void kernel_launch(void* const* d_in, const int* in_sizes, int n_in,
                              void* d_out, int out_size, void* d_ws, size_t ws_size,
                              hipStream_t stream) {
  (void)in_sizes; (void)n_in; (void)out_size; (void)ws_size;
  const float* x  = (const float*)d_in[0];
  const float* W1 = (const float*)d_in[1];
  const float* b1 = (const float*)d_in[2];
  const float* W2 = (const float*)d_in[3];
  const float* b2 = (const float*)d_in[4];
  const float* Wg = (const float*)d_in[5];
  const float* bg = (const float*)d_in[6];
  float* out  = (float*)d_out;
  float* mask = out + (size_t)NROWS * DIM;

  char* ws = (char*)d_ws;
  float*  gate = (float*)(ws);                      // 128 KB
  int*    idx  = (int*)(ws + 0x20000);              // 6.5 KB (padded to KPAD)
  float*  h    = (float*)(ws + 0x30000);            // 13.6 MB fp32 [KPAD, DIM]
  __bf16* w1p  = (__bf16*)(ws + 0xE00000);          // 8 MB packed bf16
  __bf16* w2p  = (__bf16*)(ws + 0x1600000);         // 8 MB packed bf16

  pack_w_kernel<<<NT16 * KT * 32 / 256, 256, 0, stream>>>(W1, w1p);
  pack_w_kernel<<<NT16 * KT * 32 / 256, 256, 0, stream>>>(W2, w2p);
  zero_out_kernel<<<2048, 256, 0, stream>>>((float4*)out, (NROWS * DIM + NROWS) / 4);
  gate_kernel<<<NROWS / 8, 256, 0, stream>>>(x, Wg, bg, gate);
  select_kernel<<<1, 1024, 0, stream>>>(gate, idx, mask);

  const int gemm_blocks = (KPAD / 64) * 32 / 8;     // 832 waves / 8 per block = 104
  gemm_kernel<0><<<gemm_blocks, 256, 0, stream>>>(x, idx, w1p, b1, h);
  gemm_kernel<1><<<gemm_blocks, 256, 0, stream>>>(h, idx, w2p, b2, out);
}
